// DIMKT_15152644620342
// MI455X (gfx1250) — matrix operand order repaired and checked by reading, not compiled, not run on hardware
//
#include <hip/hip_runtime.h>

// ---------------------------------------------------------------------------
// DIMKT on MI455X (gfx1250): bf16 WMMA for all GEMMs, f32 accumulation.
//   Phase 0: convert weights f32 -> bf16 (workspace, L2-resident)
//   Phase 1: t-parallel precompute GEMMs (x_all, pre_sdf1/2, pre_pka1/2, pre_ki)
//            (bf16-A GEMMs stage their A tile via GLOBAL_LOAD_ASYNC_TO_LDS)
//   Phase 2: 499-step recurrence, 16 independent workgroups (one per 16-row
//            batch tile); the 160KB of recurrent weights are staged into LDS
//            by the Tensor Data Mover (tensor_load_to_lds with HW row padding).
// ---------------------------------------------------------------------------

typedef __bf16 bf16;
typedef __attribute__((ext_vector_type(16))) __bf16 v16bf;
typedef __attribute__((ext_vector_type(8)))  __bf16 v8bf;
typedef __attribute__((ext_vector_type(8)))  float  v8f;
typedef __attribute__((ext_vector_type(4)))  unsigned int u32x4;
typedef __attribute__((ext_vector_type(8)))  int i32x8;
typedef __attribute__((ext_vector_type(4)))  int i32x4;

constexpr int NB = 256;             // batch
constexpr int NS = 500;             // sequence
constexpr int ND = 128;             // hidden dim
constexpr int NM = NB * NS;         // 128000 GEMM rows
constexpr int LP = ND + 8;          // padded LDS row pitch (136 elems, 16B-aligned rows)

// Build a 16x(32-slice) bf16 fragment from two 16-byte LDS/global chunks.
__device__ __forceinline__ v16bf ldfrag(const bf16* p, int o1, int o2) {
  v8bf lo = *reinterpret_cast<const v8bf*>(p + o1);
  v8bf hi = *reinterpret_cast<const v8bf*>(p + o2);
  return __builtin_shufflevector(lo, hi, 0,1,2,3,4,5,6,7,8,9,10,11,12,13,14,15);
}

__device__ __forceinline__ v8f wmma_bf16(v16bf a, v16bf b, v8f c) {
  // (neg_a, A, neg_b, B, c_mod, C, reuse_a, reuse_b)
  return __builtin_amdgcn_wmma_f32_16x16x32_bf16(false, a, false, b, (short)0, c,
                                                 false, false);
}

__device__ __forceinline__ float sigm(float x) { return 1.0f / (1.0f + __expf(-x)); }
__device__ __forceinline__ float tanh_f(float x) {
  float e = __expf(-2.0f * x);
  return (1.0f - e) / (1.0f + e);
}

// TDM: 2D tile load global->LDS, optional per-row padding. Arity differs
// between ROCm 7.2 (5 args) and therock (6 args); the therock toolchain is the
// only one shipping the gfx1250 TDM header, so key off that.
__device__ __forceinline__ void tdm_load_2d(u32x4 g0, i32x8 g1) {
  i32x4 z4 = {0, 0, 0, 0};
#if __has_include(<hip/amd_detail/amd_gfx1250_TDM.h>)
  i32x8 z8 = {0, 0, 0, 0, 0, 0, 0, 0};
  __builtin_amdgcn_tensor_load_to_lds(g0, g1, z4, z4, z8, 0);
#else
  __builtin_amdgcn_tensor_load_to_lds(g0, g1, z4, z4, 0);
#endif
}

// ---------------------------------------------------------------------------
// Phase 0: f32 -> bf16 weight conversion (optionally a column slice).
// ---------------------------------------------------------------------------
__global__ void wconv_kernel(const float* __restrict__ src, int stride, int col0,
                             int rows, int cols, bf16* __restrict__ dst) {
  int idx = blockIdx.x * 256 + threadIdx.x;
  if (idx < rows * cols) {
    int r = idx / cols, c = idx % cols;
    dst[idx] = (bf16)src[r * stride + col0 + c];
  }
}

// ---------------------------------------------------------------------------
// Phase 1: generic batched GEMM  out[M,128] = A[M,K] * W[128,K]^T + bias
//   A = concat of up to 4 f32 [M,128] tensors along K (a0..a3), or one bf16
//       [M,K] tensor (abf != nullptr; staged with async-to-LDS, ASYNCcnt).
//   W = bf16 row-major [128][K] (rows are output features -> B fragments).
//   One workgroup = 64 M-rows; 8 waves = (4 M-subtiles) x (2 N-halves).
// ---------------------------------------------------------------------------
__global__ __launch_bounds__(256) void gemm_pre_kernel(
    const float* __restrict__ a0, const float* __restrict__ a1,
    const float* __restrict__ a2, const float* __restrict__ a3,
    const bf16* __restrict__ abf, int K,
    const bf16* __restrict__ Wb, const float* __restrict__ bias,
    bf16* __restrict__ out) {
  __shared__ bf16 a_lds[64 * (512 + 8)];   // 66.5 KB, max K = 512
  const int tid = threadIdx.x;
  const int m0  = blockIdx.x * 64;
  const int pitch = K + 8;

  if (abf) {
    // bf16 source: pure data movement -> async global-to-LDS (16B per lane).
    for (int e = tid * 8; e < 64 * K; e += 256 * 8) {
      int r = e / K, c = e % K;
      unsigned ldso = (unsigned)(size_t)(void*)&a_lds[r * pitch + c];
      unsigned long long ga =
          (unsigned long long)(size_t)(abf + (m0 + r) * K + c);
      asm volatile("global_load_async_to_lds_b128 %0, %1, off"
                   :: "v"(ldso), "v"(ga) : "memory");
    }
    asm volatile("s_wait_asynccnt 0x0" ::: "memory");
  } else {
    // f32 sources: load 32B, convert to bf16, store to LDS.
    for (int e = tid * 8; e < 64 * K; e += 256 * 8) {
      int r = e / K, c = e % K;
      bf16* dst = &a_lds[r * pitch + c];
      int srci = c >> 7, cw = c & 127;
      const float* s = (srci == 0 ? a0 : srci == 1 ? a1 : srci == 2 ? a2 : a3)
                       + (m0 + r) * ND + cw;
#pragma unroll
      for (int q = 0; q < 8; ++q) dst[q] = (bf16)s[q];
    }
  }
  __syncthreads();

  const int w = tid >> 5, lane = tid & 31, lh = lane >> 4, lm = lane & 15;
  const int msub = w & 3, nhalf = w >> 2;
  const bf16* arow = &a_lds[(msub * 16 + lm) * pitch];

  v8f acc[4] = {};
  const int K32 = K >> 5;
  for (int kt = 0; kt < K32; ++kt) {
    const int kb = kt * 32;
    v16bf a = ldfrag(arow, kb + lh * 16, kb + lh * 16 + 8);   // A row lm, contiguous k = 16*lh + s (same as B)
#pragma unroll
    for (int j = 0; j < 4; ++j) {
      int n = nhalf * 64 + j * 16 + lm;                      // B rows = out features
      v16bf b = ldfrag(Wb + n * K, kb + lh * 16, kb + lh * 16 + 8);
      acc[j] = wmma_bf16(a, b, acc[j]);
    }
  }

#pragma unroll
  for (int j = 0; j < 4; ++j) {
    int colj = nhalf * 64 + j * 16 + lm;
    float bj = bias[colj];
#pragma unroll
    for (int i = 0; i < 8; ++i) {
      int row = m0 + msub * 16 + lh * 8 + i;                 // C layout (ISA 7.12.2)
      out[row * ND + colj] = (bf16)(acc[j][i] + bj);
    }
  }
}

// ---------------------------------------------------------------------------
// Phase 2: recurrence. Grid = 16 (batch tiles of 16 rows), block = 256 (8 waves).
// Wave w owns output columns [16w,16w+16). h carried in f32 C-layout registers;
// exchanged per step via LDS as bf16 A-fragments. 20 WMMAs / step.
// Weights staged into padded LDS by the Tensor Data Mover.
// ---------------------------------------------------------------------------
__global__ __launch_bounds__(256) void recur_kernel(
    const bf16* __restrict__ xall,
    const bf16* __restrict__ pre1, const bf16* __restrict__ pre2,
    const bf16* __restrict__ prep1, const bf16* __restrict__ prep2,
    const bf16* __restrict__ preki,
    const bf16* __restrict__ wrec,     // 5 x [128][128] bf16: sdf1,sdf2,pka1a,pka2a,kia
    const float* __restrict__ h0, float* __restrict__ out) {
  __shared__ bf16 wl[5 * 128 * LP];    // 174 KB resident recurrent weights
  __shared__ bf16 h_l[16 * LP];
  __shared__ bf16 s_l[16 * LP];

  const int tid = threadIdx.x;
  const int bt  = blockIdx.x;          // batch tile
  const int w = tid >> 5, lane = tid & 31, lh = lane >> 4, lm = lane & 15;
  const int col = w * 16 + lm;         // this lane's output feature (B row)

  // --- TDM: DMA the 640x128 bf16 weight block into LDS with a 16B pad after
  // every 256B row (pad_interval=5 -> 64 DWORDs, pad_amount=3 -> 4 DWORDs),
  // reproducing the LP=136 padded layout. Wave 0 issues; TENSORcnt tracks.
  if (w == 0) {
    unsigned ldsb = (unsigned)(size_t)(void*)&wl[0];
    unsigned long long ga = (unsigned long long)(size_t)wrec;
    u32x4 g0 = { 1u,                                   // count=1 (valid, user)
                 ldsb,                                  // lds_addr
                 (unsigned)ga,                          // global_addr[31:0]
                 (unsigned)(ga >> 32) | (2u << 30) };   // global_addr[56:32]|type=2
    i32x8 g1 = { (1 << 16) | (1 << 20) | (5 << 22) | (3 << 25), // 2B, pad en/int/amt
                 128 << 16,                             // tensor_dim0 = 128
                 640 << 16,                             // tensor_dim1 = 640
                 128 << 16,                             // tile_dim0 = 128
                 640,                                   // tile_dim1 = 640
                 128,                                   // tensor_dim0_stride = 128
                 0, 0 };
    tdm_load_2d(g0, g1);
    __builtin_amdgcn_s_wait_tensorcnt(0);
  }

  // Initial h: f32 C-layout registers + bf16 LDS copy for A-fragments.
  float h[8];
#pragma unroll
  for (int i = 0; i < 8; ++i) h[i] = h0[(bt * 16 + lh * 8 + i) * ND + col];
  {
    int r = tid >> 4, j = tid & 15;
#pragma unroll
    for (int c = 0; c < 8; ++c)
      h_l[r * LP + j * 8 + c] = (bf16)h0[(bt * 16 + r) * ND + j * 8 + c];
  }

  // Prefetched per-step precomputed gate inputs (C-layout, f32 registers).
  float p1[8], p2[8], pg[8], pp1[8], pp2[8];
#pragma unroll
  for (int i = 0; i < 8; ++i) {
    int base = ((bt * 16 + lh * 8 + i) * NS + 0) * ND + col;
    p1[i] = (float)pre1[base];  p2[i] = (float)pre2[base];
    pg[i] = (float)preki[base]; pp1[i] = (float)prep1[base];
    pp2[i] = (float)prep2[base];
  }
  __syncthreads();                     // weights (TDM) + h visible to all waves

  const bf16* w1r = &wl[(0 * 128 + col) * LP];
  const bf16* w2r = &wl[(1 * 128 + col) * LP];
  const bf16* wp1 = &wl[(2 * 128 + col) * LP];
  const bf16* wp2 = &wl[(3 * 128 + col) * LP];
  const bf16* wgr = &wl[(4 * 128 + col) * LP];

  for (int t = 0; t < NS - 1; ++t) {
    // --- stage 1: sdf1/sdf2/gamma GEMMs from h ---
    v8f a1v = {}, a2v = {}, agv = {};
#pragma unroll
    for (int kt = 0; kt < 4; ++kt) {
      const int kb = kt * 32;
      v16bf ha = ldfrag(&h_l[lm * LP], kb + lh * 16, kb + lh * 16 + 8);
      a1v = wmma_bf16(ha, ldfrag(w1r, kb + lh * 16, kb + lh * 16 + 8), a1v);
      a2v = wmma_bf16(ha, ldfrag(w2r, kb + lh * 16, kb + lh * 16 + 8), a2v);
      agv = wmma_bf16(ha, ldfrag(wgr, kb + lh * 16, kb + lh * 16 + 8), agv);
    }
    float gam[8];
#pragma unroll
    for (int i = 0; i < 8; ++i) {
      float u1 = p1[i] - a1v[i];                     // (x_t - h) W1^T + b1
      float u2 = p2[i] - a2v[i];
      float sdf = sigm(u1) * tanh_f(u2);
      gam[i] = sigm(agv[i] + pg[i]);
      s_l[(lh * 8 + i) * LP + col] = (bf16)sdf;
    }
    __syncthreads();                                 // sdf tile visible

    // --- stage 2: pka GEMMs from sdf ---
    v8f b1v = {}, b2v = {};
#pragma unroll
    for (int kt = 0; kt < 4; ++kt) {
      const int kb = kt * 32;
      v16bf sa = ldfrag(&s_l[lm * LP], kb + lh * 16, kb + lh * 16 + 8);
      b1v = wmma_bf16(sa, ldfrag(wp1, kb + lh * 16, kb + lh * 16 + 8), b1v);
      b2v = wmma_bf16(sa, ldfrag(wp2, kb + lh * 16, kb + lh * 16 + 8), b2v);
    }
#pragma unroll
    for (int i = 0; i < 8; ++i) {
      float pka = sigm(b1v[i] + pp1[i]) * tanh_f(b2v[i] + pp2[i]);
      h[i] = gam[i] * h[i] + (1.0f - gam[i]) * pka;
      h_l[(lh * 8 + i) * LP + col] = (bf16)h[i];
    }
    // Software-prefetch next step's gate inputs (independent of the barrier).
    if (t < NS - 2) {
#pragma unroll
      for (int i = 0; i < 8; ++i) {
        int base = ((bt * 16 + lh * 8 + i) * NS + (t + 1)) * ND + col;
        p1[i] = (float)pre1[base];  p2[i] = (float)pre2[base];
        pg[i] = (float)preki[base]; pp1[i] = (float)prep1[base];
        pp2[i] = (float)prep2[base];
      }
    }
    __syncthreads();                                 // h tile visible

    // --- y_t = sigmoid(<x_{t+1}, h>) : coalesced dot + wave32 shfl reduce ---
    {
      int r = tid >> 4, j = tid & 15;
      const bf16* xr = xall + ((bt * 16 + r) * NS + (t + 1)) * ND + j * 8;
      const bf16* hr = &h_l[r * LP + j * 8];
      float part = 0.0f;
#pragma unroll
      for (int c = 0; c < 8; ++c) part += (float)hr[c] * (float)xr[c];
      part += __shfl_xor(part, 1);
      part += __shfl_xor(part, 2);
      part += __shfl_xor(part, 4);
      part += __shfl_xor(part, 8);
      if ((tid & 15) == 0) out[(bt * 16 + r) * (NS - 1) + t] = sigm(part);
    }
  }
}

// ---------------------------------------------------------------------------
extern "C" void kernel_launch(void* const* d_in, const int* in_sizes, int n_in,
                              void* d_out, int out_size, void* d_ws, size_t ws_size,
                              hipStream_t stream) {
  const float* q_emb  = (const float*)d_in[0];
  const float* c_emb  = (const float*)d_in[1];
  const float* qd_emb = (const float*)d_in[2];
  const float* cd_emb = (const float*)d_in[3];
  const float* corr   = (const float*)d_in[4];
  const float* h0     = (const float*)d_in[5];
  const float* Wx     = (const float*)d_in[6];
  const float* bx     = (const float*)d_in[7];
  const float* Wsdf1  = (const float*)d_in[8];
  const float* bsdf1  = (const float*)d_in[9];
  const float* Wsdf2  = (const float*)d_in[10];
  const float* bsdf2  = (const float*)d_in[11];
  const float* Wpka1  = (const float*)d_in[12];
  const float* bpka1  = (const float*)d_in[13];
  const float* Wpka2  = (const float*)d_in[14];
  const float* bpka2  = (const float*)d_in[15];
  const float* Wki    = (const float*)d_in[16];
  const float* bki    = (const float*)d_in[17];
  float* out = (float*)d_out;

  // Workspace layout (bf16): 6 M x D tensors + converted weights (~197 MB).
  const size_t TE = (size_t)NM * ND;               // 16,384,000 elems
  bf16* xall  = (bf16*)d_ws;
  bf16* pre1  = xall + TE;
  bf16* pre2  = pre1 + TE;
  bf16* prep1 = pre2 + TE;
  bf16* prep2 = prep1 + TE;
  bf16* preki = prep2 + TE;
  bf16* wrec  = preki + TE;                        // 5 x 128 x 128
  bf16* wx    = wrec + 5 * 128 * 128;              // 128 x 512
  bf16* wp1b  = wx   + 128 * 512;                  // 128 x 128 (W_pka1[:,128:])
  bf16* wp2b  = wp1b + 128 * 128;
  bf16* wkib  = wp2b + 128 * 128;                  // 128 x 384 (W_ki[:,128:])

  // Phase 0: weight conversion.
  auto wc = [&](const float* s, int stride, int col0, int rows, int cols, bf16* d) {
    int n = rows * cols;
    wconv_kernel<<<dim3((n + 255) / 256), dim3(256), 0, stream>>>(s, stride, col0,
                                                                  rows, cols, d);
  };
  wc(Wsdf1, 128, 0, 128, 128, wrec + 0 * 16384);
  wc(Wsdf2, 128, 0, 128, 128, wrec + 1 * 16384);
  wc(Wpka1, 256, 0, 128, 128, wrec + 2 * 16384);   // sdf half
  wc(Wpka2, 256, 0, 128, 128, wrec + 3 * 16384);
  wc(Wki,   512, 0, 128, 128, wrec + 4 * 16384);   // h half
  wc(Wx,    512, 0, 128, 512, wx);
  wc(Wpka1, 256, 128, 128, 128, wp1b);             // ct half
  wc(Wpka2, 256, 128, 128, 128, wp2b);
  wc(Wki,   512, 128, 128, 384, wkib);             // [ct,qd,cd] part

  // Phase 1: t-parallel precompute GEMMs (2000 blocks x 256 threads each).
  const dim3 gg(NM / 64), gb(256);
  gemm_pre_kernel<<<gg, gb, 0, stream>>>(q_emb, c_emb, qd_emb, cd_emb, nullptr,
                                         512, wx, bx, xall);
  gemm_pre_kernel<<<gg, gb, 0, stream>>>(nullptr, nullptr, nullptr, nullptr, xall,
                                         128, wrec + 0 * 16384, bsdf1, pre1);
  gemm_pre_kernel<<<gg, gb, 0, stream>>>(nullptr, nullptr, nullptr, nullptr, xall,
                                         128, wrec + 1 * 16384, bsdf2, pre2);
  gemm_pre_kernel<<<gg, gb, 0, stream>>>(corr, nullptr, nullptr, nullptr, nullptr,
                                         128, wp1b, bpka1, prep1);
  gemm_pre_kernel<<<gg, gb, 0, stream>>>(corr, nullptr, nullptr, nullptr, nullptr,
                                         128, wp2b, bpka2, prep2);
  gemm_pre_kernel<<<gg, gb, 0, stream>>>(corr, qd_emb, cd_emb, nullptr, nullptr,
                                         384, wkib, bki, preki);

  // Phase 2: recurrence (16 independent batch tiles).
  recur_kernel<<<dim3(NB / 16), dim3(256), 0, stream>>>(xall, pre1, pre2, prep1,
                                                        prep2, preki, wrec, h0, out);
}
